// HeteroLayer_33578054320529
// MI455X (gfx1250) — compile-verified, matrix operand order checked by reading, not run on hardware
//
// MI455X (gfx1250) fused dense-GAT, 2 layers.
// Layer: Wh=h@W (WMMA f16->f32), f_src/f_dst dots, then single-pass fused
// masked-exp softmax + attn@Wh (WMMA f16->f32) streaming adj non-temporally.
#include <hip/hip_runtime.h>

typedef __attribute__((ext_vector_type(16))) _Float16 v16h;
typedef __attribute__((ext_vector_type(8)))  _Float16 v8h;
typedef __attribute__((ext_vector_type(8)))  float    v8f;
typedef __attribute__((ext_vector_type(4)))  float    v4f;

#define NROW   8192
#define NHID   256
#define GALPHA 0.2f
#define ECLAMP 10.0f   // exp clamp for f16 safety; applied to num & den consistently
#define PROW   72      // p-tile LDS row stride in halves (144B): bank-conflict-free

__device__ __forceinline__ v16h cat8(v8h lo, v8h hi) {
  return __builtin_shufflevector(lo, hi, 0,1,2,3,4,5,6,7,8,9,10,11,12,13,14,15);
}

__device__ __forceinline__ v16h cvt16(v4f a0, v4f a1, v4f a2, v4f a3) {
  v16h r;
  r[0]=(_Float16)a0[0]; r[1]=(_Float16)a0[1]; r[2]=(_Float16)a0[2]; r[3]=(_Float16)a0[3];
  r[4]=(_Float16)a1[0]; r[5]=(_Float16)a1[1]; r[6]=(_Float16)a1[2]; r[7]=(_Float16)a1[3];
  r[8]=(_Float16)a2[0]; r[9]=(_Float16)a2[1]; r[10]=(_Float16)a2[2]; r[11]=(_Float16)a2[3];
  r[12]=(_Float16)a3[0]; r[13]=(_Float16)a3[1]; r[14]=(_Float16)a3[2]; r[15]=(_Float16)a3[3];
  return r;
}

// W [K,256] f32 -> Wt [256,K] f16 (N-major, K-contiguous: WMMA B-operand ready)
__global__ __launch_bounds__(256)
void transpose_w_kernel(const float* __restrict__ W, _Float16* __restrict__ Wt, int K) {
  int idx = blockIdx.x * 256 + threadIdx.x;   // idx < K*256
  int k = idx >> 8;
  int n = idx & 255;
  Wt[(size_t)n * K + k] = (_Float16)W[idx];
}

// Wh = h @ W.  h:[8192,K] f32, Wt:[256,K] f16.  Emits Wh f16 row-major and WhT f16.
// Block 128 thr (4 waves): 64 rows x 64 cols per block; wave = 16 rows x 64 cols.
template<int K>
__global__ __launch_bounds__(128)
void gemm_wh_kernel(const float* __restrict__ h, const _Float16* __restrict__ Wt,
                    _Float16* __restrict__ Wh, _Float16* __restrict__ WhT) {
  const int lane  = threadIdx.x & 31;
  const int wave  = threadIdx.x >> 5;
  const int iBase = blockIdx.x * 64 + wave * 16;
  const int nBase = blockIdx.y * 64;
  const int mrow  = iBase + (lane & 15);
  const int kb    = (lane >> 4) * 8;           // K sub-offset per WMMA 16-bit layout

  v8f acc[4] = {};
  const float* ap = h + (size_t)mrow * K + kb;

  for (int k0 = 0; k0 < K; k0 += 32) {
    v4f a0 = *(const v4f*)(ap + k0);
    v4f a1 = *(const v4f*)(ap + k0 + 4);
    v4f a2 = *(const v4f*)(ap + k0 + 16);
    v4f a3 = *(const v4f*)(ap + k0 + 20);
    v16h af = cvt16(a0, a1, a2, a3);
#pragma unroll
    for (int nt = 0; nt < 4; nt++) {
      const _Float16* bp = Wt + (size_t)(nBase + nt * 16 + (lane & 15)) * K + k0 + kb;
      v16h bf = cat8(*(const v8h*)bp, *(const v8h*)(bp + 16));
      acc[nt] = __builtin_amdgcn_wmma_f32_16x16x32_f16(false, af, false, bf,
                                                       (short)0, acc[nt], false, false);
    }
  }

  const int iOff = (lane >> 4) * 8;            // C layout: lanes 16-31 hold M=r+8
#pragma unroll
  for (int nt = 0; nt < 4; nt++) {
    const int f = nBase + nt * 16 + (lane & 15);
#pragma unroll
    for (int r = 0; r < 8; r++) {
      const int i = iBase + r + iOff;
      _Float16 v = (_Float16)acc[nt][r];
      Wh [(size_t)i * NHID + f] = v;
      WhT[(size_t)f * NROW + i] = v;
    }
  }
}

// f_src[i] = Wh[i,:].a_src ; f_dst[i] = Wh[i,:].a_dst.  One wave per row.
__global__ __launch_bounds__(256)
void fsrc_fdst_kernel(const _Float16* __restrict__ Wh, const float* __restrict__ a,
                      float* __restrict__ fsrc, float* __restrict__ fdst) {
  const int lane = threadIdx.x & 31;
  const int row  = blockIdx.x * 8 + (threadIdx.x >> 5);
  v8h w = *(const v8h*)(Wh + (size_t)row * NHID + lane * 8);
  v4f s0 = *(const v4f*)(a + lane * 8);
  v4f s1 = *(const v4f*)(a + lane * 8 + 4);
  v4f d0 = *(const v4f*)(a + NHID + lane * 8);
  v4f d1 = *(const v4f*)(a + NHID + lane * 8 + 4);
  float fs = 0.f, fd = 0.f;
#pragma unroll
  for (int q = 0; q < 4; q++) {
    float w0 = (float)w[q], w1 = (float)w[q + 4];
    fs += w0 * s0[q] + w1 * s1[q];
    fd += w0 * d0[q] + w1 * d1[q];
  }
#pragma unroll
  for (int m = 16; m >= 1; m >>= 1) {
    fs += __shfl_xor(fs, m, 32);
    fd += __shfl_xor(fd, m, 32);
  }
  if (lane == 0) { fsrc[row] = fs; fdst[row] = fd; }
}

// Fused: p_ij = adj>0 ? exp(min(leakyrelu(fs_i+fd_j),10)) : 0 ; out_i = elu(elu((p@Wh)/sum_j p)).
// Block = 16 rows, 4 waves; wave w owns features [64w,64w+64).
// Single pass over adj; 16x64 p tile per barrier (8 WMMAs/wave per sync), double-buffered.
__global__ __launch_bounds__(128)
void gat_attn_kernel(const float* __restrict__ adj, const float* __restrict__ fsrc,
                     const float* __restrict__ fdst, const _Float16* __restrict__ WhT,
                     float* __restrict__ out) {
  __shared__ __align__(16) _Float16 pbuf[2][16 * PROW];  // 2 x 2.25 KB
  __shared__ float dsum[16];

  const int t    = threadIdx.x;
  const int lane = t & 31;
  const int wave = t >> 5;
  const int i0   = blockIdx.x * 16;
  const int pr   = t >> 3;          // producer row   0..15
  const int pj   = (t & 7) * 8;     // producer j-off 0..56

  if (t < 16) dsum[t] = 0.0f;
  __syncthreads();

  const float  fs   = fsrc[i0 + pr];
  const float* adjp = adj + (size_t)(i0 + pr) * NROW;
  float dacc = 0.0f;

  v8f acc[4] = {};
  const int arow = lane & 15;
  const int kb   = (lane >> 4) * 8;
  const int fblk = wave * 64;

  int buf = 0;
  for (int j0 = 0; j0 < NROW; j0 += 64, buf ^= 1) {
    // ---- produce 16x64 p tile (f16) into LDS; stream adj non-temporally ----
    v4f av0 = __builtin_nontemporal_load((const v4f*)(adjp + j0 + pj));
    v4f av1 = __builtin_nontemporal_load((const v4f*)(adjp + j0 + pj + 4));
    v4f fv0 = *(const v4f*)(fdst + j0 + pj);
    v4f fv1 = *(const v4f*)(fdst + j0 + pj + 4);
    if (j0 + 64 < NROW) __builtin_prefetch(adjp + j0 + 64 + pj, 0, 3);
    v8h pv;
#pragma unroll
    for (int q = 0; q < 8; q++) {
      float e = fs + ((q < 4) ? fv0[q & 3] : fv1[q & 3]);
      e = (e > 0.f) ? e : GALPHA * e;       // leaky_relu
      e = fminf(e, ECLAMP);
      float aq = (q < 4) ? av0[q & 3] : av1[q & 3];
      float p = (aq > 0.f) ? __expf(e) : 0.f;
      dacc += p;
      pv[q] = (_Float16)p;
    }
    *(v8h*)&pbuf[buf][pr * PROW + pj] = pv;   // one aligned 16B LDS store
    __syncthreads();

    // ---- consume: 2 A frags from LDS, 8 WMMAs against L2-resident WhT ----
    const _Float16* plb = &pbuf[buf][arow * PROW + kb];
#pragma unroll
    for (int ks = 0; ks < 64; ks += 32) {
      const _Float16* pl = plb + ks;
      v16h af = cat8(*(const v8h*)pl, *(const v8h*)(pl + 16));
#pragma unroll
      for (int nt = 0; nt < 4; nt++) {
        const _Float16* bp = WhT + (size_t)(fblk + nt * 16 + arow) * NROW + j0 + ks + kb;
        v16h bf = cat8(*(const v8h*)bp, *(const v8h*)(bp + 16));
        acc[nt] = __builtin_amdgcn_wmma_f32_16x16x32_f16(false, af, false, bf,
                                                         (short)0, acc[nt], false, false);
      }
    }
  }

  atomicAdd(&dsum[pr], dacc);     // ds_add_f32, 8 partials per row
  __syncthreads();
  if (t < 16) dsum[t] = 1.0f / dsum[t];   // invert once; epilogue multiplies
  __syncthreads();

  const int iOff = (lane >> 4) * 8;
#pragma unroll
  for (int nt = 0; nt < 4; nt++) {
    const int f = fblk + nt * 16 + arow;
#pragma unroll
    for (int r = 0; r < 8; r++) {
      const int il = r + iOff;
      float v = acc[nt][r] * dsum[il];
      v = (v > 0.f) ? v : (__expf(v) - 1.0f);   // elu (layer)
      v = (v > 0.f) ? v : (__expf(v) - 1.0f);   // elu (outer)
      out[(size_t)(i0 + il) * NHID + f] = v;
    }
  }
}

extern "C" void kernel_launch(void* const* d_in, const int* in_sizes, int n_in,
                              void* d_out, int out_size, void* d_ws, size_t ws_size,
                              hipStream_t stream) {
  const float* x   = (const float*)d_in[0];   // [8192,512]
  const float* adj = (const float*)d_in[1];   // [8192,8192]
  const float* W1  = (const float*)d_in[2];   // [512,256]
  const float* a1  = (const float*)d_in[3];   // [512,1]
  const float* W2  = (const float*)d_in[4];   // [256,256]
  const float* a2  = (const float*)d_in[5];   // [512,1]
  float* out = (float*)d_out;

  char* ws = (char*)d_ws;
  _Float16* Wh   = (_Float16*)(ws);                              // 4 MB
  _Float16* WhT  = (_Float16*)(ws + (4u << 20));                 // 4 MB
  _Float16* Wt   = (_Float16*)(ws + (8u << 20));                 // 256 KB max
  float*    fsrc = (float*)   (ws + (8u << 20) + (512u << 10));  // 32 KB
  float*    fdst = fsrc + NROW;                                  // 32 KB
  float*    h1   = (float*)   (ws + (9u << 20));                 // 8 MB

  dim3 gGemm(NROW / 64, NHID / 64);

  // ---------------- layer 1 (K = 512) ----------------
  transpose_w_kernel<<<512, 256, 0, stream>>>(W1, Wt, 512);
  gemm_wh_kernel<512><<<gGemm, 128, 0, stream>>>(x, Wt, Wh, WhT);
  fsrc_fdst_kernel<<<NROW / 8, 256, 0, stream>>>(Wh, a1, fsrc, fdst);
  gat_attn_kernel<<<NROW / 16, 128, 0, stream>>>(adj, fsrc, fdst, WhT, h1);

  // ---------------- layer 2 (K = 256) ----------------
  transpose_w_kernel<<<256, 256, 0, stream>>>(W2, Wt, 256);
  gemm_wh_kernel<256><<<gGemm, 128, 0, stream>>>(h1, Wt, Wh, WhT);
  fsrc_fdst_kernel<<<NROW / 8, 256, 0, stream>>>(Wh, a2, fsrc, fdst);
  gat_attn_kernel<<<NROW / 16, 128, 0, stream>>>(adj, fsrc, fdst, WhT, out);
}
